// DisentangledTransformer_3504693313932
// MI455X (gfx1250) — matrix operand order repaired and checked by reading, not compiled, not run on hardware
//
#include <hip/hip_runtime.h>
#include <hip/hip_bf16.h>
#include <stdint.h>

// ---------------------------------------------------------------------------
// Types for CDNA5 WMMA (wave32): v_wmma_f32_16x16x32_bf16
// ---------------------------------------------------------------------------
typedef __attribute__((ext_vector_type(16))) __bf16 v16bf;
typedef __attribute__((ext_vector_type(8)))  float  v8f;

union FragBf { v16bf v; uint4 q[2]; };

__device__ __forceinline__ __bf16 f2bf(float f) {
  unsigned u = __builtin_bit_cast(unsigned, f);
  u += 0x7FFFu + ((u >> 16) & 1u);            // round-to-nearest-even
  unsigned short h = (unsigned short)(u >> 16);
  return __builtin_bit_cast(__bf16, h);
}

// CDNA5 async global->LDS DMA (ASYNCcnt-tracked). IOFFSET is applied to both
// the LDS and global address, so one base pair serves 4 consecutive B128s.
__device__ __forceinline__ void async_fill_row64B(const void* lds_ptr, const void* gptr) {
  unsigned l = (unsigned)(uintptr_t)lds_ptr;   // LDS addr = generic addr[31:0]
  asm volatile(
      "global_load_async_to_lds_b128 %0, %1, off\n\t"
      "global_load_async_to_lds_b128 %0, %1, off offset:16\n\t"
      "global_load_async_to_lds_b128 %0, %1, off offset:32\n\t"
      "global_load_async_to_lds_b128 %0, %1, off offset:48"
      :: "v"(l), "v"(gptr) : "memory");
}
__device__ __forceinline__ void async_fill_16B(const void* lds_ptr, const void* gptr) {
  unsigned l = (unsigned)(uintptr_t)lds_ptr;
  asm volatile("global_load_async_to_lds_b128 %0, %1, off"
               :: "v"(l), "v"(gptr) : "memory");
}
__device__ __forceinline__ void wait_async0() {
  asm volatile("s_wait_asynccnt 0x0" ::: "memory");
}

// ---------------------------------------------------------------------------
// Tiled bf16 GEMM (TN):  C[M,N] = A[M,K] * Bt[N,K]^T
// block = 256 threads (8 wave32), tile 256x64, wave tile 64x32 (4x2 WMMA),
// K-step 32, double-buffered LDS with async-DMA pipeline.
// grid = (N/64, M/256, batch).
// ---------------------------------------------------------------------------
template<bool OUT_BF16>
__global__ __launch_bounds__(256)
void gemm_tn_kernel(const __bf16* __restrict__ A,
                    const __bf16* __restrict__ Bt,
                    void* __restrict__ Cv,
                    int K, int lda, int ldb, int ldc,
                    long long sA, long long sB, long long sC)
{
  __shared__ __align__(16) __bf16 As[2][256][40];  // [buf][m][k], 80B pitch
  __shared__ __align__(16) __bf16 Bs[2][64][40];   // [buf][n][k], 80B pitch

  const int tid  = threadIdx.x;
  const int lane = tid & 31;
  const int wave = tid >> 5;
  const int wm   = wave >> 1;   // 0..3 -> 64-row M sub-tile
  const int wn   = wave & 1;    // 0..1 -> 32-col N sub-tile

  const long long z = blockIdx.z;
  A  += z * sA;
  Bt += z * sB;

  const int m0 = blockIdx.y * 256;
  const int n0 = blockIdx.x * 64;

  v8f acc[4][2];
  #pragma unroll
  for (int i = 0; i < 4; ++i)
    #pragma unroll
    for (int j = 0; j < 2; ++j)
      #pragma unroll
      for (int r = 0; r < 8; ++r) acc[i][j][r] = 0.0f;

  // DMA mapping: A -> one 64B row per thread; B -> one B128 per thread
  const int am = tid;                  // 0..255
  const int bn = tid >> 2;             // 0..63
  const int bc = (tid & 3) * 8;        // 0,8,16,24

  // fragment read mapping (CDNA5 16-bit WMMA VGPR layouts)
  const int arow0 = wm * 64 + (lane & 15);
  const int akoff = (lane >> 4) * 8;    // lanes 0-15: K0-7/16-23; 16-31: 8-15/24-31
  const int bcol0 = wn * 32 + (lane & 15);
  const int bkoff = (lane >> 4) * 8;    // lanes 0-15: K0-7/16-23; 16-31: 8-15/24-31

  const __bf16* srcA0 = A  + (size_t)(m0 + am) * lda;
  const __bf16* srcB0 = Bt + (size_t)(n0 + bn) * ldb + bc;

  const int nk = K >> 5;               // K / 32

  // prologue: stage tile 0 into buffer 0
  async_fill_row64B(&As[0][am][0], srcA0);
  async_fill_16B(&Bs[0][bn][bc], srcB0);
  wait_async0();
  __syncthreads();

  for (int s = 0; s < nk; ++s) {
    const int cur = s & 1;
    if (s + 1 < nk) {                  // stage next tile into other buffer
      const int k1 = (s + 1) << 5;
      async_fill_row64B(&As[cur ^ 1][am][0], srcA0 + k1);
      async_fill_16B(&Bs[cur ^ 1][bn][bc], srcB0 + k1);
      if (s + 2 < nk)
        __builtin_prefetch(srcA0 + k1 + 32, 0, 0);  // global_prefetch_b8
    }

    FragBf fa[4], fb[2];
    #pragma unroll
    for (int mi = 0; mi < 4; ++mi) {
      const __bf16* p = &As[cur][arow0 + mi * 16][akoff];
      fa[mi].q[0] = *(const uint4*)(p);
      fa[mi].q[1] = *(const uint4*)(p + 16);
    }
    #pragma unroll
    for (int ni = 0; ni < 2; ++ni) {
      const __bf16* p = &Bs[cur][bcol0 + ni * 16][bkoff];
      fb[ni].q[0] = *(const uint4*)(p);
      fb[ni].q[1] = *(const uint4*)(p + 16);
    }
    #pragma unroll
    for (int mi = 0; mi < 4; ++mi)
      #pragma unroll
      for (int ni = 0; ni < 2; ++ni)
        acc[mi][ni] = __builtin_amdgcn_wmma_f32_16x16x32_bf16(
            false, fa[mi].v, false, fb[ni].v, (short)0, acc[mi][ni],
            false, false);

    wait_async0();                     // next tile's DMA landed (per wave)
    __syncthreads();                   // everyone done reading cur + DMA done
  }

  // ---- store C (f32 layout: VGPR r -> M=r (lanes 0-15) / M=r+8 (16-31)) ----
  const int cn0 = n0 + wn * 32 + (lane & 15);
  const int cm0 = m0 + wm * 64 + (lane >> 4) * 8;
  #pragma unroll
  for (int mi = 0; mi < 4; ++mi)
    #pragma unroll
    for (int ni = 0; ni < 2; ++ni) {
      const int n  = cn0 + ni * 16;
      const int mB = cm0 + mi * 16;
      #pragma unroll
      for (int r = 0; r < 8; ++r) {
        float v = acc[mi][ni][r];
        size_t idx = (size_t)z * sC + (size_t)(mB + r) * ldc + n;
        if (OUT_BF16) ((__bf16*)Cv)[idx] = f2bf(v);
        else          ((float*)Cv)[idx]  = v;
      }
    }
}

// ---------------------------------------------------------------------------
// Tiled transposes (coalesced both sides).  block (32,8), 32x32 tiles.
// ---------------------------------------------------------------------------
__global__ __launch_bounds__(256)
void transpose_f32_bf16_kernel(const float* __restrict__ src,
                               __bf16* __restrict__ dst,
                               int R, int C, long long sS, long long sD)
{
  __shared__ float tile[32][33];
  const int c0 = blockIdx.x * 32, r0 = blockIdx.y * 32;
  src += (long long)blockIdx.z * sS;
  dst += (long long)blockIdx.z * sD;
  const int tx = threadIdx.x, ty = threadIdx.y;
  #pragma unroll
  for (int i = 0; i < 32; i += 8)
    tile[ty + i][tx] = src[(size_t)(r0 + ty + i) * C + c0 + tx];
  __syncthreads();
  #pragma unroll
  for (int i = 0; i < 32; i += 8)
    dst[(size_t)(c0 + ty + i) * R + r0 + tx] = f2bf(tile[tx][ty + i]);
}

__global__ __launch_bounds__(256)
void transpose_bf16_kernel(const __bf16* __restrict__ src,
                           __bf16* __restrict__ dst,
                           int R, int C, long long sS, long long sD)
{
  __shared__ __bf16 tile[32][33];
  const int c0 = blockIdx.x * 32, r0 = blockIdx.y * 32;
  src += (long long)blockIdx.z * sS;
  dst += (long long)blockIdx.z * sD;
  const int tx = threadIdx.x, ty = threadIdx.y;
  #pragma unroll
  for (int i = 0; i < 32; i += 8)
    tile[ty + i][tx] = src[(size_t)(r0 + ty + i) * C + c0 + tx];
  __syncthreads();
  #pragma unroll
  for (int i = 0; i < 32; i += 8)
    dst[(size_t)(c0 + ty + i) * R + r0 + tx] = tile[tx][ty + i];
}

// ---------------------------------------------------------------------------
// h0 = [x | onehot(t)] in bf16.  i over B*T*320.
// ---------------------------------------------------------------------------
__global__ void build_h0_kernel(const float* __restrict__ x,
                                __bf16* __restrict__ h0, long long total)
{
  long long i = (long long)blockIdx.x * blockDim.x + threadIdx.x;
  if (i >= total) return;
  int d = (int)(i % 320);
  long long r = i / 320;          // r = b*T + t
  int t = (int)(r & 255);
  float v = (d < 64) ? x[r * 64 + d] : ((d - 64) == t ? 1.0f : 0.0f);
  h0[i] = f2bf(v);
}

__global__ void f32_to_bf16_kernel(const float* __restrict__ in,
                                   __bf16* __restrict__ out, long long n)
{
  long long i = (long long)blockIdx.x * blockDim.x + threadIdx.x;
  if (i < n) out[i] = f2bf(in[i]);
}

// strided row copy (concat prefix): dst[r*ldd + c] = src[r*lds + c]
__global__ void copy_rows_kernel(const __bf16* __restrict__ src,
                                 __bf16* __restrict__ dst,
                                 int cols, int lds, int ldd, long long total)
{
  long long i = (long long)blockIdx.x * blockDim.x + threadIdx.x;
  if (i >= total) return;
  long long r = i / cols;
  int c = (int)(i % cols);
  dst[r * (long long)ldd + c] = src[r * (long long)lds + c];
}

// ---------------------------------------------------------------------------
// Causal softmax over rows of [B*T, 256] f32 scores -> bf16 probs
// ---------------------------------------------------------------------------
__global__ __launch_bounds__(256)
void softmax_causal_kernel(const float* __restrict__ S,
                           __bf16* __restrict__ P)
{
  __shared__ float red[256];
  const int r = blockIdx.x;
  const int t = r & 255;
  const int s = threadIdx.x;
  const size_t base = (size_t)r * 256;

  float v = (s <= t) ? S[base + s] : -3.4e38f;
  red[s] = v;
  __syncthreads();
  #pragma unroll
  for (int off = 128; off > 0; off >>= 1) {
    if (s < off) red[s] = fmaxf(red[s], red[s + off]);
    __syncthreads();
  }
  const float mx = red[0];
  __syncthreads();

  float e = (s <= t) ? __expf(v - mx) : 0.0f;
  red[s] = e;
  __syncthreads();
  #pragma unroll
  for (int off = 128; off > 0; off >>= 1) {
    if (s < off) red[s] += red[s + off];
    __syncthreads();
  }
  const float inv = 1.0f / red[0];
  P[base + s] = f2bf(e * inv);
}

// ---------------------------------------------------------------------------
// Host orchestration
// ---------------------------------------------------------------------------
extern "C" void kernel_launch(void* const* d_in, const int* in_sizes, int n_in,
                              void* d_out, int out_size, void* d_ws, size_t ws_size,
                              hipStream_t stream)
{
  (void)in_sizes; (void)n_in; (void)out_size; (void)ws_size;

  const float* x  = (const float*)d_in[0];
  const float* A0 = (const float*)d_in[1];
  const float* A1 = (const float*)d_in[2];
  const float* Wo = (const float*)d_in[3];

  constexpr int Bsz = 64, T = 256;
  constexpr int D0 = 320, D1 = 960, D2 = 2880;
  constexpr long long BT = (long long)Bsz * T;   // 16384

  size_t off = 0;
  auto alloc = [&](size_t bytes) -> char* {
    char* p = (char*)d_ws + off;
    off = (off + bytes + 255) & ~(size_t)255;
    return p;
  };
  __bf16* h0   = (__bf16*)alloc(BT * D0 * 2);          // 10.5 MB
  __bf16* h0T  = (__bf16*)alloc(BT * D0 * 2);          // 10.5 MB  [b][d][t]
  __bf16* h1   = (__bf16*)alloc(BT * D1 * 2);          // 31.5 MB
  __bf16* h1T  = (__bf16*)alloc(BT * D1 * 2);          // 31.5 MB  [b][d][t]
  __bf16* h2   = (__bf16*)alloc(BT * D2 * 2);          // 94.4 MB
  __bf16* G    = (__bf16*)alloc(BT * D1 * 2);          // 31.5 MB (max width)
  float*  Sc   = (float*) alloc(BT * T * 4);           // 16.8 MB (one head)
  __bf16* P    = (__bf16*)alloc(BT * T * 2);           //  8.4 MB
  __bf16* A0bT = (__bf16*)alloc((size_t)2 * D0 * D0 * 2);  // [h][e][d]
  __bf16* A1bT = (__bf16*)alloc((size_t)2 * D1 * D1 * 2);  // [h][e][d]
  __bf16* Wob  = (__bf16*)alloc((size_t)64 * D2 * 2);      // already [N][K]

  const dim3 blk(256);
  const dim3 tblk(32, 8);

  // weights: convert (+ transpose A0/A1 so every GEMM uses the TN fast path)
  transpose_f32_bf16_kernel<<<dim3(10, 10, 2), tblk, 0, stream>>>(
      A0, A0bT, D0, D0, (long long)D0 * D0, (long long)D0 * D0);
  transpose_f32_bf16_kernel<<<dim3(30, 30, 2), tblk, 0, stream>>>(
      A1, A1bT, D1, D1, (long long)D1 * D1, (long long)D1 * D1);
  {
    long long n = 64LL * D2;
    f32_to_bf16_kernel<<<dim3((unsigned)((n + 255) / 256)), blk, 0, stream>>>(Wo, Wob, n);
  }
  // h0 = [x | onehot] ; h0T = h0^T per batch
  {
    long long n = BT * D0;
    build_h0_kernel<<<dim3((unsigned)((n + 255) / 256)), blk, 0, stream>>>(x, h0, n);
  }
  transpose_bf16_kernel<<<dim3(10, 8, 64), tblk, 0, stream>>>(
      h0, h0T, T, D0, (long long)T * D0, (long long)T * D0);
  // h1 prefix = h0
  {
    long long n = BT * D0;
    copy_rows_kernel<<<dim3((unsigned)((n + 255) / 256)), blk, 0, stream>>>(
        h0, h1, D0, D0, D1, n);
  }

  // ---------------- layer 0 (d = 320) ----------------
  for (int head = 0; head < 2; ++head) {
    // G = h0 @ A0[head] : Bt = A0^T [e][d]
    gemm_tn_kernel<true><<<dim3(5, 1, 64), blk, 0, stream>>>(
        h0, A0bT + (size_t)head * D0 * D0, G,
        D0, D0, D0, D0, (long long)T * D0, 0, (long long)T * D0);
    // Sc = G @ h0^T : Bt = h0 [s][e]
    gemm_tn_kernel<false><<<dim3(4, 1, 64), blk, 0, stream>>>(
        G, h0, Sc,
        D0, D0, D0, T, (long long)T * D0, (long long)T * D0, (long long)T * T);
    softmax_causal_kernel<<<dim3((unsigned)BT), blk, 0, stream>>>(Sc, P);
    // h1[:, :, 320+head*320] = P @ h0 : Bt = h0T [d][s]
    gemm_tn_kernel<true><<<dim3(5, 1, 64), blk, 0, stream>>>(
        P, h0T, h1 + D0 + head * D0,
        T, T, T, D1, (long long)T * T, (long long)T * D0, (long long)T * D1);
  }

  // h2 prefix = h1 ; h1T = h1^T per batch
  {
    long long n = BT * D1;
    copy_rows_kernel<<<dim3((unsigned)((n + 255) / 256)), blk, 0, stream>>>(
        h1, h2, D1, D1, D2, n);
  }
  transpose_bf16_kernel<<<dim3(30, 8, 64), tblk, 0, stream>>>(
      h1, h1T, T, D1, (long long)T * D1, (long long)T * D1);

  // ---------------- layer 1 (d = 960) ----------------
  for (int head = 0; head < 2; ++head) {
    gemm_tn_kernel<true><<<dim3(15, 1, 64), blk, 0, stream>>>(
        h1, A1bT + (size_t)head * D1 * D1, G,
        D1, D1, D1, D1, (long long)T * D1, 0, (long long)T * D1);
    gemm_tn_kernel<false><<<dim3(4, 1, 64), blk, 0, stream>>>(
        G, h1, Sc,
        D1, D1, D1, T, (long long)T * D1, (long long)T * D1, (long long)T * T);
    softmax_causal_kernel<<<dim3((unsigned)BT), blk, 0, stream>>>(Sc, P);
    gemm_tn_kernel<true><<<dim3(15, 1, 64), blk, 0, stream>>>(
        P, h1T, h2 + D1 + head * D1,
        T, T, T, D2, (long long)T * T, (long long)T * D1, (long long)T * D2);
  }

  // ---------------- output projection ----------------
  // logits[BT, 64] = h2 @ Wo^T : f32 output
  gemm_tn_kernel<false><<<dim3(1, 64, 1), blk, 0, stream>>>(
      h2, Wob, (float*)d_out,
      D2, D2, D2, 64, 0, 0, 0);
}
